// TMPNN_26388279067204
// MI455X (gfx1250) — matrix operand order repaired and checked by reading, not compiled, not run on hardware
//
#include <hip/hip_runtime.h>
#include <hip/hip_bf16.h>

// ---------------------------------------------------------------------------
// TaylorMap polynomial network, CDNA5 (gfx1250) wave32 + WMMA bf16.
//
// Per step:  x <- x + bias + X@W1 + quad,  quad[b,d] = sum_{i,j} x_i x_j W2[i,j,d]
// Treated as a K=16512 GEMM whose A rows (x_i * x_j) are generated in-register.
// ---------------------------------------------------------------------------

typedef __attribute__((ext_vector_type(16))) __bf16 v16bf;
typedef __attribute__((ext_vector_type(8)))  __bf16 v8bf;
typedef __attribute__((ext_vector_type(8)))  float  v8f;
typedef __attribute__((ext_vector_type(4)))  int    v4i;

#define N_FEATURES 120
#define R_TARGETS  4
#define LATENT     4
#define Dm         128            // padded state dim
#define BATCH      16384
#define STEPS      7

#define KK_TOTAL   16512          // W rows 1..16512 (linear 128 + quadratic 16384)
#define NCHUNKS    (KK_TOTAL / 32)        // 516 K-chunks of 32
#define NTILES     (Dm / 16)              // 8 N-tiles of 16
#define TILE_ELEMS 512                    // one 32x16 bf16 B tile (1 KB)
#define CHUNK_ELEMS (TILE_ELEMS * NTILES) // 4096 bf16 per K-chunk
#define SLAB_CHUNKS 4                     // one slab = 4 chunks = one quadratic i
#define NSLABS     (NCHUNKS / SLAB_CHUNKS)          // 129 (slab 0 = linear term)
#define SLAB_ELEMS (CHUNK_ELEMS * SLAB_CHUNKS)      // 16384 bf16
#define SLAB_BYTES (SLAB_ELEMS * 2)                 // 32 KB

#define WG_THREADS 128
#define WAVES      4
#define ROWS_PER_WG 64
#define XSTRIDE    136                    // bf16 elems; 272B row stride (bank spread)
#define XS_BYTES   (ROWS_PER_WG * XSTRIDE * 2)      // 17408
#define LDS_BYTES  (XS_BYTES + 2 * SLAB_BYTES)      // 82944 (dynamic LDS; <320KB/WGP)
#define ASYNC_PER_WAVE (SLAB_BYTES / (WG_THREADS * 16))  // 16 async b128 per wave/slab

#if __has_builtin(__builtin_amdgcn_global_load_async_to_lds_b128) && \
    __has_builtin(__builtin_amdgcn_s_wait_asynccnt)
#define USE_ASYNC 1
#else
#define USE_ASYNC 0
#endif

// ---------------------------------------------------------------------------
// Pre-pass: W[1..16512][128] fp32 -> bf16, pre-swizzled into WMMA B-operand
// lane layout.  B tile (kc,nt) covers K=kc*32+kl, N=nt*16+nl;
//   lane = (kl>>4)*16 + nl, element e = kl&15  (32B contiguous per lane).
// ---------------------------------------------------------------------------
__global__ void wswizzle_kernel(const float* __restrict__ W,
                                __bf16* __restrict__ wb) {
  int idx = blockIdx.x * blockDim.x + threadIdx.x;
  if (idx >= KK_TOTAL * Dm) return;
  int kk = idx >> 7;            // 0..16511  (W row = kk+1)
  int d  = idx & 127;
  int kc = kk >> 5, kl = kk & 31;
  int nt = d >> 4,  nl = d & 15;
  int lane = ((kl >> 4) << 4) | nl;
  int e    = kl & 15;
  size_t dst = (size_t)(kc * NTILES + nt) * TILE_ELEMS + lane * 16 + e;
  wb[dst] = (__bf16)W[(size_t)(1 + kk) * Dm + d];
}

// Stage one 32KB slab of swizzled W into an LDS buffer (all 128 threads).
__device__ __forceinline__ void stage_slab(const __bf16* __restrict__ wb,
                                           __bf16* buf, int s, int tid) {
  const char* g = (const char*)wb + (size_t)s * SLAB_BYTES;
  char* l = (char*)buf;
#if USE_ASYNC
  #pragma unroll
  for (int p = 0; p < ASYNC_PER_WAVE; ++p) {
    int off = (p * WG_THREADS + tid) * 16;
    __builtin_amdgcn_global_load_async_to_lds_b128(
        (__attribute__((address_space(1))) v4i*)(g + off),
        (__attribute__((address_space(3))) v4i*)(l + off),
        0, 0);
  }
#else
  #pragma unroll
  for (int p = 0; p < ASYNC_PER_WAVE; ++p) {
    int off = (p * WG_THREADS + tid) * 16;
    *(uint4*)(l + off) = *(const uint4*)(g + off);
  }
#endif
}

// ---------------------------------------------------------------------------
// Main kernel: 4 waves x 16 rows per WG; state kept in the 8 f32 WMMA
// accumulators (C layout) across all 7 steps.
// ---------------------------------------------------------------------------
__global__ __launch_bounds__(WG_THREADS) void taylor_map_kernel(
    const float* __restrict__ X, const float* __restrict__ W,
    const float* __restrict__ t0, const float* __restrict__ l0,
    const __bf16* __restrict__ wb, float* __restrict__ out) {
  extern __shared__ __attribute__((aligned(128))) char smem[];
  __bf16* xs    = (__bf16*)smem;                 // [64][136] bf16 state copy
  __bf16* slab0 = (__bf16*)(smem + XS_BYTES);
  __bf16* slab1 = slab0 + SLAB_ELEMS;

  const int tid  = threadIdx.x;
  const int lane = tid & 31;
  const int wave = tid >> 5;
  const int half = lane >> 4;       // 0/1 lane half
  const int n    = lane & 15;       // N index (C layout) / M row (A layout)
  const int rowBase = blockIdx.x * ROWS_PER_WG + wave * 16;
  const int lrow0   = wave * 16;    // this wave's row block in xs

  // --- init: bias (W row 0, step-invariant) + state x0 in C layout ---------
  float bias[NTILES];
  v8f acc[NTILES];
  #pragma unroll
  for (int nt = 0; nt < NTILES; ++nt) {
    const int col = nt * 16 + n;
    bias[nt] = W[col];
    #pragma unroll
    for (int r = 0; r < 8; ++r) {
      const int row = rowBase + r + 8 * half;   // C layout: vgpr r, lane half
      float v;
      if (col < N_FEATURES)                 v = X[(size_t)row * N_FEATURES + col];
      else if (col < N_FEATURES + R_TARGETS) v = t0[col - N_FEATURES];
      else                                   v = l0[col - N_FEATURES - R_TARGETS];
      acc[nt][r] = v;
    }
  }

  for (int step = 0; step < STEPS; ++step) {
    // 1) publish fp32 state to LDS as bf16 (per-wave rows only)
    #pragma unroll
    for (int nt = 0; nt < NTILES; ++nt)
      #pragma unroll
      for (int r = 0; r < 8; ++r)
        xs[(lrow0 + r + 8 * half) * XSTRIDE + nt * 16 + n] = (__bf16)acc[nt][r];
    __syncthreads();

    // 2) A-layout base tiles of x (16x32 bf16, 4 K-chunks).  Per lane (M=n):
    //    elems 0..15 = K=jc*32+half*16.. (contiguous, matches swizzled B)
    const int arow = lrow0 + n;
    v16bf abase[4];
    #pragma unroll
    for (int jc = 0; jc < 4; ++jc) {
      const v8bf lo = *(const v8bf*)&xs[arow * XSTRIDE + jc * 32 + half * 16];
      const v8bf hi = *(const v8bf*)&xs[arow * XSTRIDE + jc * 32 + 8 + half * 16];
      #pragma unroll
      for (int e = 0; e < 8; ++e) { abase[jc][e] = lo[e]; abase[jc][8 + e] = hi[e]; }
    }

    // 3) stream W through LDS, double-buffered; slab 0 = linear term (A=x),
    //    slab s>=1 = quadratic block i=s-1 (A = x_i * x).
    stage_slab(wb, slab0, 0, tid);
    for (int s = 0; s < NSLABS; ++s) {
      __bf16* buf  = (s & 1) ? slab1 : slab0;
      __bf16* nbuf = (s & 1) ? slab0 : slab1;
      if (s + 1 < NSLABS) stage_slab(wb, nbuf, s + 1, tid);
#if USE_ASYNC
      if (s + 1 < NSLABS) __builtin_amdgcn_s_wait_asynccnt(ASYNC_PER_WAVE);
      else                __builtin_amdgcn_s_wait_asynccnt(0);
#endif
      __syncthreads();

      if (s == 0) {
        #pragma unroll
        for (int jc = 0; jc < 4; ++jc) {
          #pragma unroll
          for (int nt = 0; nt < NTILES; ++nt) {
            const v16bf b =
                *(const v16bf*)&buf[(jc * NTILES + nt) * TILE_ELEMS + lane * 16];
            acc[nt] = __builtin_amdgcn_wmma_f32_16x16x32_bf16(
                false, abase[jc], false, b, (short)0, acc[nt], false, false);
          }
        }
      } else {
        const __bf16 xi = xs[arow * XSTRIDE + (s - 1)];   // x[row, i], bf16
        #pragma unroll
        for (int jc = 0; jc < 4; ++jc) {
          v16bf a;                      // x_i * x[j-chunk]  (v_pk_mul_bf16)
          #pragma unroll
          for (int e = 0; e < 16; ++e) a[e] = (__bf16)(xi * abase[jc][e]);
          #pragma unroll
          for (int nt = 0; nt < NTILES; ++nt) {
            const v16bf b =
                *(const v16bf*)&buf[(jc * NTILES + nt) * TILE_ELEMS + lane * 16];
            acc[nt] = __builtin_amdgcn_wmma_f32_16x16x32_bf16(
                false, a, false, b, (short)0, acc[nt], false, false);
          }
        }
      }
      __syncthreads();   // free `buf` for the slab issued two iterations ahead
    }

    // 4) bias (residual is implicit: C operand started as x)
    #pragma unroll
    for (int nt = 0; nt < NTILES; ++nt)
      #pragma unroll
      for (int r = 0; r < 8; ++r) acc[nt][r] += bias[nt];
  }

  // 5) output cols 120..123 -> N-tile 7, n in [8,12)
  if (n >= 8 && n < 12) {
    #pragma unroll
    for (int r = 0; r < 8; ++r)
      out[(size_t)(rowBase + r + 8 * half) * R_TARGETS + (n - 8)] = acc[7][r];
  }
}

// ---------------------------------------------------------------------------
extern "C" void kernel_launch(void* const* d_in, const int* in_sizes, int n_in,
                              void* d_out, int out_size, void* d_ws, size_t ws_size,
                              hipStream_t stream) {
  const float* X  = (const float*)d_in[0];   // [16384,120]
  const float* W  = (const float*)d_in[1];   // [16513,128]
  const float* t0 = (const float*)d_in[2];   // [4]
  const float* l0 = (const float*)d_in[3];   // [4]
  float* out = (float*)d_out;                // [16384,4]
  __bf16* wb = (__bf16*)d_ws;                // 4,227,072 bytes of swizzled bf16 W

  (void)in_sizes; (void)n_in; (void)out_size; (void)ws_size;

  {
    const int total = KK_TOTAL * Dm;
    wswizzle_kernel<<<total / 256, 256, 0, stream>>>(W, wb);
  }

  (void)hipFuncSetAttribute(reinterpret_cast<const void*>(taylor_map_kernel),
                            hipFuncAttributeMaxDynamicSharedMemorySize, LDS_BYTES);
  taylor_map_kernel<<<BATCH / ROWS_PER_WG, WG_THREADS, LDS_BYTES, stream>>>(
      X, W, t0, l0, wb, out);
}
